// Seq2SeqModel_86337432584447
// MI455X (gfx1250) — matrix operand order repaired and checked by reading, not compiled, not run on hardware
//
#include <hip/hip_runtime.h>
#include <hip/hip_bf16.h>

typedef __attribute__((ext_vector_type(16))) _Float16 v16h;
typedef __attribute__((ext_vector_type(8)))  float    v8f;

#define WAVES 4
#define TILE  16
#define BSZ   8192
#define TENC  64
#define TDEC  10

// ---- fragment index helpers (CDNA5 16-bit WMMA layouts, wave32) ----
__device__ __forceinline__ int ka_idx(int lane, int e) {
    int base = (lane & 16) ? 8 : 0;
    return base + ((e < 8) ? e : (8 + e));
}
__device__ __forceinline__ int kb_idx(int lane, int e) {
    int base = (lane & 16) ? 8 : 0;
    return base + ((e < 8) ? e : (8 + e));
}

#if __has_builtin(__builtin_amdgcn_tanhf)
__device__ __forceinline__ float tanhf_(float x) { return __builtin_amdgcn_tanhf(x); }
__device__ __forceinline__ float sigf(float x)   { return __builtin_fmaf(0.5f, __builtin_amdgcn_tanhf(0.5f * x), 0.5f); }
#elif __has_builtin(__builtin_amdgcn_tanh_f32)
__device__ __forceinline__ float tanhf_(float x) { return __builtin_amdgcn_tanh_f32(x); }
__device__ __forceinline__ float sigf(float x)   { return __builtin_fmaf(0.5f, __builtin_amdgcn_tanh_f32(0.5f * x), 0.5f); }
#else
__device__ __forceinline__ float fast_exp(float x) { return __builtin_amdgcn_exp2f(x * 1.44269504f); }
__device__ __forceinline__ float sigf(float x)  { return __builtin_amdgcn_rcpf(1.0f + fast_exp(-x)); }
__device__ __forceinline__ float tanhf_(float x){ return 1.0f - 2.0f * __builtin_amdgcn_rcpf(fast_exp(2.0f * x) + 1.0f); }
#endif

__device__ __forceinline__ v8f wmma16(v16h a, v16h b, v8f c) {
    return __builtin_amdgcn_wmma_f32_16x16x32_f16(false, a, false, b, (short)0, c, false, false);
}

__device__ __forceinline__ v16h zero16h() {
    v16h z;
#pragma unroll
    for (int e = 0; e < 16; ++e) z[e] = (_Float16)0.0f;
    return z;
}
__device__ __forceinline__ v8f zero8f() {
    v8f z;
#pragma unroll
    for (int r = 0; r < 8; ++r) z[r] = 0.0f;
    return z;
}

// =====================================================================
// Fused 4-layer encoder: wave w == layer w, skewed pipeline t = cyc - w.
// Layer handoff + own recurrence via double-buffered LDS fragments
// (parity = t&1), one __syncthreads per cycle. Weights in per-wave
// registers. X tile pre-staged to LDS so no global access sits on the
// barrier-synchronized critical path.
// =====================================================================
__global__ __launch_bounds__(WAVES * 32)
void encoder_fused_kernel(const float* __restrict__ X,       // [B,T,6] f32
                          const float* __restrict__ wih0, const float* __restrict__ whh0,
                          const float* __restrict__ bih0, const float* __restrict__ bhh0,
                          const float* __restrict__ wih1, const float* __restrict__ whh1,
                          const float* __restrict__ bih1, const float* __restrict__ bhh1,
                          const float* __restrict__ wih2, const float* __restrict__ whh2,
                          const float* __restrict__ bih2, const float* __restrict__ bhh2,
                          const float* __restrict__ wih3, const float* __restrict__ whh3,
                          const float* __restrict__ bih3, const float* __restrict__ bhh3,
                          _Float16* __restrict__ hTs,        // [4][B/16][32][16] frag layout
                          float* __restrict__ cTs)           // [4][B][32] f32 C-tile order
{
    __shared__ __attribute__((aligned(32))) _Float16 hfrag[WAVES][2][32][16]; // 8KB
    __shared__ __attribute__((aligned(32))) _Float16 xlds[TILE][TENC * 6];    // 12KB

    const int tid  = threadIdx.x;
    const int wave = tid >> 5, lane = tid & 31;
    const int bt = blockIdx.x;                 // one 16-row tile per block
    const int rowbase = bt * TILE;
    const int n = lane & 15;
    const int mrowoff = (lane < 16) ? 0 : 8;

    // ---- cooperative X tile stage (contiguous span, coalesced, f32->f16) ----
    {
        const float* xsrc = X + (size_t)rowbase * (TENC * 6);
        _Float16* xdst = &xlds[0][0];
        for (int idx = tid; idx < TILE * TENC * 6; idx += WAVES * 32)
            xdst[idx] = (_Float16)xsrc[idx];
    }

    // select this wave's layer parameters (branch chains avoid scratch arrays)
    const float* Wih = (wave == 0) ? wih0 : (wave == 1) ? wih1 : (wave == 2) ? wih2 : wih3;
    const float* Whh = (wave == 0) ? whh0 : (wave == 1) ? whh1 : (wave == 2) ? whh2 : whh3;
    const float* bih = (wave == 0) ? bih0 : (wave == 1) ? bih1 : (wave == 2) ? bih2 : bih3;
    const float* bhh = (wave == 0) ? bhh0 : (wave == 1) ? bhh1 : (wave == 2) ? bhh2 : bhh3;
    const int I = (wave == 0) ? 6 : 32;

    // stage this layer's weight fragments into registers (one-time gather)
    v16h Wi[8], Wh[8];
#pragma unroll
    for (int nt = 0; nt < 8; ++nt) {
#pragma unroll
        for (int e = 0; e < 16; ++e) {
            int k = kb_idx(lane, e);
            Wi[nt][e] = (k < I) ? (_Float16)Wih[(nt * 16 + n) * I + k] : (_Float16)0.0f;
            Wh[nt][e] = (_Float16)Whh[(nt * 16 + n) * 32 + k];
        }
    }
    float biasv[8];
#pragma unroll
    for (int nt = 0; nt < 8; ++nt) biasv[nt] = bih[nt * 16 + n] + bhh[nt * 16 + n];

    // zero both parities of this wave's h buffer, then sync before pipeline
    *(v16h*)&hfrag[wave][0][lane][0] = zero16h();
    *(v16h*)&hfrag[wave][1][lane][0] = zero16h();
    __syncthreads();

    float cst[2][8];
#pragma unroll
    for (int jt = 0; jt < 2; ++jt)
#pragma unroll
        for (int r = 0; r < 8; ++r) cst[jt][r] = 0.0f;

    const v8f ZC = zero8f();

    // wave0 input fragment: elements 6..15 (and all of lanes>=16) stay zero
    v16h Ax = zero16h();

#pragma unroll 1
    for (int cyc = 0; cyc < TENC + WAVES - 1; ++cyc) {
        const int t = cyc - wave;
        if (t >= 0 && t < TENC) {
            // ---- input fragment ----
            if (wave == 0) {
                if (lane < 16) {
#pragma unroll
                    for (int e = 0; e < 6; ++e) Ax[e] = xlds[n][t * 6 + e];
                }
            } else {
                Ax = *(const v16h*)&hfrag[wave - 1][t & 1][lane][0];
            }
            // ---- own h_{t-1} (parity (t-1)&1; zero-initialized for t=0) ----
            v16h Ah = *(const v16h*)&hfrag[wave][(t - 1) & 1][lane][0];

#pragma unroll
            for (int jt = 0; jt < 2; ++jt) {
                v8f acc[4];   // i, f, g, o tiles for this j-half
#pragma unroll
                for (int q = 0; q < 4; ++q) {
                    v8f a = wmma16(Ax, Wi[q * 2 + jt], ZC);
                    acc[q] = wmma16(Ah, Wh[q * 2 + jt], a);
                }
                float bi = biasv[jt], bf = biasv[2 + jt], bg = biasv[4 + jt], bo = biasv[6 + jt];
#pragma unroll
                for (int r = 0; r < 8; ++r) {
                    float cn = sigf(acc[1][r] + bf) * cst[jt][r]
                             + sigf(acc[0][r] + bi) * tanhf_(acc[2][r] + bg);
                    cst[jt][r] = cn;
                    float hn = sigf(acc[3][r] + bo) * tanhf_(cn);
                    int L = (r + mrowoff) + ((n & 8) ? 16 : 0);
                    int e = jt * 8 + (n & 7);
                    hfrag[wave][t & 1][L][e] = (_Float16)hn;
                }
            }
        }
        __syncthreads();   // cycle boundary: publish h(t) to downstream wave
    }

    // ---- epilogue: final h (fragment layout) + final c (C-tile order) ----
    v16h hlast = *(const v16h*)&hfrag[wave][(TENC - 1) & 1][lane][0];
    *(v16h*)(hTs + (((size_t)wave * (BSZ / TILE) + bt) * 32 + lane) * 16) = hlast;
    float* cT = cTs + (size_t)wave * BSZ * 32;
#pragma unroll
    for (int jt = 0; jt < 2; ++jt)
#pragma unroll
        for (int r = 0; r < 8; ++r)
            cT[(size_t)(rowbase + r + mrowoff) * 32 + jt * 16 + n] = cst[jt][r];
}

// =====================================================================
// Decoder: 10 steps x 4 cells; encoder states constant across steps.
// =====================================================================
__global__ __launch_bounds__(WAVES * 32)
void decoder_kernel(const float* __restrict__ dec_in,                    // [B,6]
                    const _Float16* __restrict__ h0s, const _Float16* __restrict__ h1s,
                    const _Float16* __restrict__ h2s, const _Float16* __restrict__ h3s,
                    const float* __restrict__ c0s, const float* __restrict__ c1s,
                    const float* __restrict__ c2s, const float* __restrict__ c3s,
                    const float* __restrict__ wih0, const float* __restrict__ whh0,
                    const float* __restrict__ bih0, const float* __restrict__ bhh0,
                    const float* __restrict__ wih1, const float* __restrict__ whh1,
                    const float* __restrict__ bih1, const float* __restrict__ bhh1,
                    const float* __restrict__ wih2, const float* __restrict__ whh2,
                    const float* __restrict__ bih2, const float* __restrict__ bhh2,
                    const float* __restrict__ wih3, const float* __restrict__ whh3,
                    const float* __restrict__ bih3, const float* __restrict__ bhh3,
                    const float* __restrict__ denseW, const float* __restrict__ denseB,
                    float* __restrict__ out)                              // [B,10,6]
{
    __shared__ __attribute__((aligned(32))) _Float16 wlds[4][2][8][32][16];
    __shared__ __attribute__((aligned(32))) _Float16 hscr[WAVES][32][16];
    __shared__ __attribute__((aligned(32))) _Float16 oscr[WAVES][32][16];

    const int tid  = threadIdx.x;
    const int wave = tid >> 5, lane = tid & 31;
    const int bt = blockIdx.x * WAVES + wave;
    const int rowbase = bt * TILE;
    const int n = lane & 15;
    const int mrowoff = (lane < 16) ? 0 : 8;

    const float* Wihs[4] = {wih0, wih1, wih2, wih3};
    const float* Whhs[4] = {whh0, whh1, whh2, whh3};
    const float* bihs[4] = {bih0, bih1, bih2, bih3};
    const float* bhhs[4] = {bhh0, bhh1, bhh2, bhh3};
    const _Float16* hs[4] = {h0s, h1s, h2s, h3s};
    const float* cs[4]   = {c0s, c1s, c2s, c3s};

    // stage all 4 cells' weight fragments to LDS (64KB)
    for (int idx = tid; idx < 4 * 2 * 8 * 32 * 16; idx += WAVES * 32) {
        int e = idx & 15, ln = (idx >> 4) & 31, frag = (idx >> 9) & 7;
        int mat = (idx >> 12) & 1, cell = idx >> 13;
        int nn = ln & 15, k = kb_idx(ln, e);
        const float* W = (mat == 0) ? Wihs[cell] : Whhs[cell];
        int K = (mat == 0 && cell == 0) ? 6 : 32;
        float v = (k < K) ? W[(frag * 16 + nn) * K + k] : 0.0f;
        wlds[cell][mat][frag][ln][e] = (_Float16)v;
    }
    __syncthreads();

    // zero-fill output fragment scratch once (only e<6 of lanes<16 ever rewritten)
    *(v16h*)&oscr[wave][lane][0] = zero16h();

    // encoder-state A fragments (contiguous, frag layout) + c state
    v16h Ah[4];
    float cenc[4][2][8];
#pragma unroll
    for (int cell = 0; cell < 4; ++cell) {
        Ah[cell] = *(const v16h*)(hs[cell] + ((size_t)bt * 32 + lane) * 16);
        const float* cp = cs[cell];
#pragma unroll
        for (int jt = 0; jt < 2; ++jt)
#pragma unroll
            for (int r = 0; r < 8; ++r)
                cenc[cell][jt][r] = cp[(size_t)(rowbase + r + mrowoff) * 32 + jt * 16 + n];
    }
    float biasv[4][8];
#pragma unroll
    for (int cell = 0; cell < 4; ++cell)
#pragma unroll
        for (int nt = 0; nt < 8; ++nt)
            biasv[cell][nt] = bihs[cell][nt * 16 + n] + bhhs[cell][nt * 16 + n];

    // dense layer B fragment (32x16, only N<6 valid)
    v16h Bd;
#pragma unroll
    for (int e = 0; e < 16; ++e) {
        int k = kb_idx(lane, e);
        Bd[e] = (n < 6) ? (_Float16)denseW[n * 32 + k] : (_Float16)0.0f;
    }
    float dbias = (n < 6) ? denseB[n] : 0.0f;

    const v8f ZC = zero8f();

    // initial x fragment from decoder_input [B,6] (one-time scattered gather)
    v16h Ax;
#pragma unroll
    for (int e = 0; e < 16; ++e) {
        int k = ka_idx(lane, e);
        Ax[e] = (k < 6) ? (_Float16)dec_in[(size_t)(rowbase + n) * 6 + k] : (_Float16)0.0f;
    }

#pragma unroll 1
    for (int step = 0; step < TDEC; ++step) {
        v16h xf = Ax;
#pragma unroll
        for (int cell = 0; cell < 4; ++cell) {
#pragma unroll
            for (int jt = 0; jt < 2; ++jt) {
                v8f acc[4];
#pragma unroll
                for (int q = 0; q < 4; ++q) {
                    v8f a = wmma16(Ah[cell], *(const v16h*)&wlds[cell][1][q * 2 + jt][lane][0], ZC);
                    acc[q] = wmma16(xf, *(const v16h*)&wlds[cell][0][q * 2 + jt][lane][0], a);
                }
                float bi = biasv[cell][jt],     bf = biasv[cell][2 + jt];
                float bg = biasv[cell][4 + jt], bo = biasv[cell][6 + jt];
#pragma unroll
                for (int r = 0; r < 8; ++r) {
                    float cn = sigf(acc[1][r] + bf) * cenc[cell][jt][r]
                             + sigf(acc[0][r] + bi) * tanhf_(acc[2][r] + bg);
                    float hn = sigf(acc[3][r] + bo) * tanhf_(cn);
                    int L = (r + mrowoff) + ((n & 8) ? 16 : 0);
                    int e = jt * 8 + (n & 7);
                    hscr[wave][L][e] = (_Float16)hn;
                }
            }
            xf = *(const v16h*)&hscr[wave][lane][0];
        }
        // dense: out = h @ Wd^T + b  (one WMMA, N padded to 16)
        v8f oacc = wmma16(xf, Bd, ZC);
#pragma unroll
        for (int r = 0; r < 8; ++r) {
            int mrow = r + mrowoff;
            if (n < 6) {
                float val = oacc[r] + dbias;
                out[((size_t)(rowbase + mrow) * TDEC + step) * 6 + n] = val;
                oscr[wave][mrow][n] = (_Float16)val;   // frag slot (L=mrow, e=n)
            }
        }
        // next step's input fragment (contiguous, zero-padded beyond k=5)
        Ax = *(const v16h*)&oscr[wave][lane][0];
    }
}

extern "C" void kernel_launch(void* const* d_in, const int* in_sizes, int n_in,
                              void* d_out, int out_size, void* d_ws, size_t ws_size,
                              hipStream_t stream) {
    const float* enc_x = (const float*)d_in[0];
    // d_in[1] = others_fut_input: unused by reference
    const float* dec_x = (const float*)d_in[2];
    // param leaves: enc1 L0 @3, enc1 L1 @7, enc2 L0 @11, enc2 L1 @15,
    //               dec1 L0 @19, dec1 L1 @23, dec2 L0 @27, dec2 L1 @31, dense @35,36
    auto P = [&](int i) { return (const float*)d_in[i]; };

    char* ws = (char*)d_ws;
    _Float16* hTs = (_Float16*)ws;                          // 4 * 512 KiB
    const size_t hstride = (size_t)BSZ * 32;                // halfs per layer
    float* cTs = (float*)(ws + 4 * hstride * sizeof(_Float16));
    const size_t cstride = (size_t)BSZ * 32;                // floats per layer

    // fused 4-layer encoder: one 16-row tile per block, 4 waves = 4 layers
    encoder_fused_kernel<<<dim3(BSZ / TILE), dim3(WAVES * 32), 0, stream>>>(
        enc_x,
        P(3),  P(4),  P(5),  P(6),
        P(7),  P(8),  P(9),  P(10),
        P(11), P(12), P(13), P(14),
        P(15), P(16), P(17), P(18),
        hTs, cTs);

    decoder_kernel<<<dim3(BSZ / (WAVES * TILE)), dim3(WAVES * 32), 0, stream>>>(
        dec_x,
        hTs + 0 * hstride, hTs + 1 * hstride, hTs + 2 * hstride, hTs + 3 * hstride,
        cTs + 0 * cstride, cTs + 1 * cstride, cTs + 2 * cstride, cTs + 3 * cstride,
        P(19), P(20), P(21), P(22),
        P(23), P(24), P(25), P(26),
        P(27), P(28), P(29), P(30),
        P(31), P(32), P(33), P(34),
        P(35), P(36), (float*)d_out);
}
